// GPNN_21449066676829
// MI455X (gfx1250) — matrix operand order repaired and checked by reading, not compiled, not run on hardware
//
#include <hip/hip_runtime.h>
#include <hip/hip_bf16.h>

// CDNA5 wave32 WMMA types
typedef __attribute__((ext_vector_type(16))) __bf16 v16bf;
typedef __attribute__((ext_vector_type(8)))  float  v8f;

union Frag16 {
    v16bf v;
    uint4 u4[2];
    unsigned short s[16];
};

// LDS pitches (in ushorts), padded for 64x4B bank spread while keeping 16B alignment
#define AP 136   // adj chunk pitch   (272B -> bank stride 68: 16 distinct banks)
#define BP 136   // alpha^T chunk pitch
#define XP 72    // x / h staging pitch (144B -> bank stride 36: 16 distinct banks)

__device__ __forceinline__ unsigned short f32_to_bf16(float f) {
    unsigned int u = __float_as_uint(f);
    u += 0x7FFFu + ((u >> 16) & 1u);   // round-to-nearest-even
    return (unsigned short)(u >> 16);
}

// gfx1250 async global->LDS copy (ASYNCcnt-tracked). Low 32 bits of a generic
// LDS pointer are the wave-relative LDS address per the flat-aperture spec.
__device__ __forceinline__ void async_copy_b128(const void* gptr, void* lptr) {
    unsigned lds_off = (unsigned)(uintptr_t)lptr;
    asm volatile("global_load_async_to_lds_b128 %0, %1, off"
                 :: "v"(lds_off), "v"(gptr) : "memory");
}
__device__ __forceinline__ void wait_async0() {
    asm volatile("s_wait_asynccnt 0x0" ::: "memory");
}

// ---------------------------------------------------------------- setup kernels

__global__ void k_zero_adj(unsigned short* __restrict__ adjb) {
    int t = blockIdx.x * blockDim.x + threadIdx.x;    // 32768 threads * 16B = 512KB
    ((uint4*)adjb)[t] = make_uint4(0u, 0u, 0u, 0u);
}

__global__ void k_scatter(const int* __restrict__ ei, unsigned short* __restrict__ adjb) {
    int e = blockIdx.x * blockDim.x + threadIdx.x;
    if (e < 8192) {
        int src = ei[e];
        int dst = ei[8192 + e];
        adjb[dst * 512 + src] = 0x3F80;               // bf16 1.0 (duplicates idempotent)
    }
}

// transpose w1,w2 (F x C, row-major) -> bf16 [n][k] so B-fragments are K-contiguous
__global__ void k_cvt_w(const float* __restrict__ w1, const float* __restrict__ w2,
                        unsigned short* __restrict__ w1T, unsigned short* __restrict__ w2T) {
    int t = blockIdx.x * blockDim.x + threadIdx.x;    // 8192 threads
    int which = t >> 12;
    int idx   = t & 4095;
    int n = idx >> 6, k = idx & 63;
    const float* w = which ? w2 : w1;
    unsigned short* o = which ? w2T : w1T;
    o[n * 64 + k] = f32_to_bf16(w[k * 64 + n]);
}

// S[v,f] = sum_j adj[v,j] * alpha[v*512+j, f]
__global__ void k_srows(const float* __restrict__ alpha,
                        const unsigned short* __restrict__ adjb,
                        float* __restrict__ Srow) {
    int v = blockIdx.x;
    int f = threadIdx.x;                              // 64 threads
    float acc = 0.0f;
    for (int j = 0; j < 512; ++j) {
        if (adjb[v * 512 + j])
            acc += alpha[(v * 512 + j) * 64 + f];
    }
    Srow[v * 64 + f] = acc;
}

// ---------------------------------------------------------------- main fused kernel
// grid = (4 v-tiles, 512 w), block = 256 (8 wave32). Each wave owns a 16x64 strip.
// GEMM1: T_w[128,64] = adj[vt*128:+128, :512] @ alpha[w*512:+512, :64]   (K=512)
// epilogue: x = T + S[v] + (1+eps)*alpha[v*512+w];  h = relu(x@w1+b1);  out = h@w2+b2

__global__ __launch_bounds__(256) void k_main(
    const float* __restrict__ alpha, const unsigned short* __restrict__ adjb,
    const unsigned short* __restrict__ w1T, const unsigned short* __restrict__ w2T,
    const float* __restrict__ b1, const float* __restrict__ b2,
    const float* __restrict__ Srow, const float* __restrict__ epsp,
    float* __restrict__ out)
{
    __shared__ unsigned short lds_a [128 * AP];  // 34KB: adj chunk [m][k]; reused as x staging [m][XP]
    __shared__ unsigned short lds_bT[128 * XP];  // 18KB: alpha chunk^T [n][BP]; reused as h staging

    const int vt   = blockIdx.x;       // 0..3
    const int w    = blockIdx.y;       // 0..511
    const int tid  = threadIdx.x;
    const int lane = tid & 31;
    const int wave = tid >> 5;
    const int l15  = lane & 15;
    const int h    = lane >> 4;        // half-wave select

    const v8f zero8 = {0.f, 0.f, 0.f, 0.f, 0.f, 0.f, 0.f, 0.f};
    v8f acc[4] = {zero8, zero8, zero8, zero8};

    const int mrow    = tid >> 1;      // 0..127 row loader
    const int halfsel = tid & 1;

    for (int kc = 0; kc < 4; ++kc) {
        // stage adj chunk (128x128 bf16) via async global->LDS (no VGPR round-trip)
        {
            const char* g = (const char*)(adjb + (vt * 128 + mrow) * 512 + kc * 128)
                          + halfsel * 128;
            char* l = (char*)(lds_a + mrow * AP) + halfsel * 128;
#pragma unroll
            for (int i = 0; i < 8; ++i)
                async_copy_b128(g + i * 16, l + i * 16);
        }
        // stage alpha chunk (128 rows x 64 f32 -> bf16, transposed to [n][k])
        {
            const float4* g = (const float4*)(alpha + (w * 512 + kc * 128 + mrow) * 64 + halfsel * 32);
            if (kc < 3)   // hint next chunk toward GL2
                __builtin_prefetch(alpha + (w * 512 + (kc + 1) * 128 + mrow) * 64, 0, 1);
#pragma unroll
            for (int i = 0; i < 8; ++i) {
                float4 fv = g[i];
                int n = halfsel * 32 + i * 4;
                lds_bT[(n + 0) * BP + mrow] = f32_to_bf16(fv.x);
                lds_bT[(n + 1) * BP + mrow] = f32_to_bf16(fv.y);
                lds_bT[(n + 2) * BP + mrow] = f32_to_bf16(fv.z);
                lds_bT[(n + 3) * BP + mrow] = f32_to_bf16(fv.w);
            }
        }
        wait_async0();
        __syncthreads();
#pragma unroll
        for (int ki = 0; ki < 4; ++ki) {
            const int k0 = ki * 32;
            // A fragment (16x32 bf16): lane half h holds K in [h*8,h*8+8) and [16+h*8,16+h*8+8)
            Frag16 afr;
            const int am = wave * 16 + l15;
            afr.u4[0] = *(const uint4*)(lds_a + am * AP + k0 + h * 8);
            afr.u4[1] = *(const uint4*)(lds_a + am * AP + k0 + 16 + h * 8);
#pragma unroll
            for (int nt = 0; nt < 4; ++nt) {
                // B fragment (32x16 bf16): lane = column n, half h holds K in [h*8,h*8+8) and [16+h*8,16+h*8+8)
                Frag16 bfr;
                const int bn = nt * 16 + l15;
                bfr.u4[0] = *(const uint4*)(lds_bT + bn * BP + k0 + h * 8);
                bfr.u4[1] = *(const uint4*)(lds_bT + bn * BP + k0 + 16 + h * 8);
                acc[nt] = __builtin_amdgcn_wmma_f32_16x16x32_bf16(
                    false, afr.v, false, bfr.v, (short)0, acc[nt], false, false);
            }
        }
        __syncthreads();
    }

    // ---- epilogue: residual + S, then fused MLP (two more bf16 WMMA GEMMs) ----
    const float scale = 1.0f + epsp[0];

#pragma unroll
    for (int nt = 0; nt < 4; ++nt) {
        const int n = nt * 16 + l15;
#pragma unroll
        for (int r = 0; r < 8; ++r) {
            const int m = wave * 16 + h * 8 + r;           // C/D layout: M = r (+8 for upper half)
            const int vg = vt * 128 + m;
            float x = acc[nt][r]
                    + Srow[vg * 64 + n]
                    + scale * alpha[(vg * 512 + w) * 64 + n];
            lds_a[m * XP + n] = f32_to_bf16(x);            // x staging [m][XP]
        }
    }
    __syncthreads();

    // GEMM2: h = relu(x @ w1 + b1), K = 64
    v8f hacc[4] = {zero8, zero8, zero8, zero8};
#pragma unroll
    for (int ks = 0; ks < 2; ++ks) {
        const int k0 = ks * 32;
        Frag16 afr;
        const int am = wave * 16 + l15;
        afr.u4[0] = *(const uint4*)(lds_a + am * XP + k0 + h * 8);
        afr.u4[1] = *(const uint4*)(lds_a + am * XP + k0 + 16 + h * 8);
#pragma unroll
        for (int nt = 0; nt < 4; ++nt) {
            Frag16 bfr;
            const int bn = nt * 16 + l15;
            bfr.u4[0] = *(const uint4*)(w1T + bn * 64 + k0 + h * 8);
            bfr.u4[1] = *(const uint4*)(w1T + bn * 64 + k0 + 16 + h * 8);
            hacc[nt] = __builtin_amdgcn_wmma_f32_16x16x32_bf16(
                false, afr.v, false, bfr.v, (short)0, hacc[nt], false, false);
        }
    }
    __syncthreads();
#pragma unroll
    for (int nt = 0; nt < 4; ++nt) {
        const int n = nt * 16 + l15;
        const float bb = b1[n];
#pragma unroll
        for (int r = 0; r < 8; ++r) {
            const int m = wave * 16 + h * 8 + r;
            float hv = hacc[nt][r] + bb;
            hv = hv > 0.f ? hv : 0.f;
            lds_bT[m * XP + n] = f32_to_bf16(hv);          // h staging [m][XP]
        }
    }
    __syncthreads();

    // GEMM3: out = h @ w2 + b2, K = 64
    v8f oacc[4] = {zero8, zero8, zero8, zero8};
#pragma unroll
    for (int ks = 0; ks < 2; ++ks) {
        const int k0 = ks * 32;
        Frag16 afr;
        const int am = wave * 16 + l15;
        afr.u4[0] = *(const uint4*)(lds_bT + am * XP + k0 + h * 8);
        afr.u4[1] = *(const uint4*)(lds_bT + am * XP + k0 + 16 + h * 8);
#pragma unroll
        for (int nt = 0; nt < 4; ++nt) {
            Frag16 bfr;
            const int bn = nt * 16 + l15;
            bfr.u4[0] = *(const uint4*)(w2T + bn * 64 + k0 + h * 8);
            bfr.u4[1] = *(const uint4*)(w2T + bn * 64 + k0 + 16 + h * 8);
            oacc[nt] = __builtin_amdgcn_wmma_f32_16x16x32_bf16(
                false, afr.v, false, bfr.v, (short)0, oacc[nt], false, false);
        }
    }
#pragma unroll
    for (int nt = 0; nt < 4; ++nt) {
        const int n = nt * 16 + l15;
        const float bb = b2[n];
#pragma unroll
        for (int r = 0; r < 8; ++r) {
            const int vg = vt * 128 + wave * 16 + h * 8 + r;
            out[(vg * 512 + w) * 64 + n] = oacc[nt][r] + bb;
        }
    }
}

// ---------------------------------------------------------------- launcher

extern "C" void kernel_launch(void* const* d_in, const int* in_sizes, int n_in,
                              void* d_out, int out_size, void* d_ws, size_t ws_size,
                              hipStream_t stream) {
    (void)in_sizes; (void)n_in; (void)out_size; (void)ws_size;
    const float* alpha = (const float*)d_in[0];
    const int*   ei    = (const int*)d_in[1];
    const float* w1    = (const float*)d_in[2];
    const float* b1    = (const float*)d_in[3];
    const float* w2    = (const float*)d_in[4];
    const float* b2    = (const float*)d_in[5];
    const float* eps   = (const float*)d_in[6];
    float* out = (float*)d_out;

    char* ws = (char*)d_ws;
    unsigned short* adjb = (unsigned short*)(ws);             // 512*512*2   = 524288 B
    unsigned short* w1T  = (unsigned short*)(ws + 524288);    // 64*64*2     =   8192 B
    unsigned short* w2T  = (unsigned short*)(ws + 532480);    //               8192 B
    float*          Srow = (float*)        (ws + 540672);     // 512*64*4    = 131072 B

    k_zero_adj<<<128, 256, 0, stream>>>(adjb);
    k_scatter <<<32, 256, 0, stream>>>(ei, adjb);
    k_cvt_w   <<<32, 256, 0, stream>>>(w1, w2, w1T, w2T);
    k_srows   <<<512, 64, 0, stream>>>(alpha, adjb, Srow);

    dim3 grid(4, 512);
    k_main<<<grid, 256, 0, stream>>>(alpha, adjb, w1T, w2T, b1, b2, Srow, eps, out);
}
